// OutlierDetector_24927990186089
// MI455X (gfx1250) — matrix operand order repaired and checked by reading, not compiled, not run on hardware
//
#include <hip/hip_runtime.h>
#include <hip/hip_bf16.h>
#include <stdint.h>

// ---- clang ext-vector types (trivially-copyable: safe inside unions) ----
typedef float    f32x4 __attribute__((ext_vector_type(4)));
typedef float    f32x2 __attribute__((ext_vector_type(2)));
typedef unsigned u32x4 __attribute__((ext_vector_type(4)));
typedef unsigned u32x2 __attribute__((ext_vector_type(2)));
typedef _Float16 v16h  __attribute__((ext_vector_type(16)));
typedef float    v8f   __attribute__((ext_vector_type(8)));

#define N_PIX 131072
#define DIM   256
#define NCLS  19
#define KPCA  50
#define KPAD  64

#define COMPS_BYTES (NCLS*DIM*KPAD*2)   // 622592 B  f16 [C][256][64] d-major
#define CINV_BYTES  (NCLS*KPAD*KPAD*2)  // 155648 B  f16 [C][64][64]

union AFrag { u32x4 q[2]; v16h h; };          // 32 B = one 16x16x32 f16 fragment / lane
union HPack { _Float16 h[4]; u32x2 u; };
union H1    { _Float16 h; unsigned short u; };

// ---- prep 1: comps f32 [C][50][256] -> f16 d-major [C][256][64], zero pad k>=50 ----
__global__ void od_prep_comps(const float* __restrict__ comps,
                              unsigned short* __restrict__ outT) {
  int flat = blockIdx.x * 256 + threadIdx.x;      // == (c*256 + d)*64 + k
  int k = flat & 63;
  int d = (flat >> 6) & 255;
  int c = flat >> 14;
  float v = 0.f;
  if (k < KPCA) v = comps[(c * KPCA + k) * DIM + d];
  H1 cv; cv.h = (_Float16)v;
  outT[flat] = cv.u;
}

// ---- prep 2: cov_inv f32 [C][50][50] -> f16 [C][64][64], zero padded ----
__global__ void od_prep_cinv(const float* __restrict__ cov_inv,
                             unsigned short* __restrict__ o) {
  int c = blockIdx.x;
  for (int i = threadIdx.x; i < KPAD * KPAD; i += 256) {
    int l = i >> 6, k = i & 63;
    float v = 0.f;
    if (l < KPCA && k < KPCA) v = cov_inv[(c * KPCA + l) * KPCA + k];
    H1 cv; cv.h = (_Float16)v;
    o[c * KPAD * KPAD + i] = cv.u;
  }
}

// ---- prep 3: fused whitening coeffs: a = rstd, b = -mean_proj*rstd - mu ----
__global__ void od_prep_ab(const float* __restrict__ mean,
                           const float* __restrict__ comps,
                           const float* __restrict__ var,
                           const float* __restrict__ mu,
                           f32x2* __restrict__ ab) {
  __shared__ float red[256];
  int b = blockIdx.x;
  int c = b >> 6, k = b & 63;
  int t = threadIdx.x;
  float p = 0.f;
  if (k < KPCA) p = mean[c * DIM + t] * comps[(c * KPCA + k) * DIM + t];
  red[t] = p;
  __syncthreads();
  for (int s = 128; s > 0; s >>= 1) {
    if (t < s) red[t] += red[t + s];
    __syncthreads();
  }
  if (t == 0) {
    f32x2 r;
    if (k < KPCA) {
      float rs = rsqrtf(var[c * KPCA + k]);
      r.x = rs;
      r.y = -red[0] * rs - mu[c * KPCA + k];
    } else { r.x = 0.f; r.y = 0.f; }
    ab[c * KPAD + k] = r;
  }
}

// ---- main: 4 waves / block, 16 rows per wave, f16 WMMA for both GEMMs ----
__global__ __launch_bounds__(128) void od_main(
    const f32x4* __restrict__ feats4,
    const unsigned short* __restrict__ compsT,  // f16 [C][256][64]
    const unsigned short* __restrict__ cinv,    // f16 [C][64][64]
    const f32x2* __restrict__ ab,               // [C][64]
    float* __restrict__ out) {
  // 64 rows x 256 cols f16, padded row stride 264 halves (528B -> conflict free)
  __shared__ unsigned short tileA[64 * 264];    // 33792 B
  __shared__ unsigned short diffB[4 * 16 * 72]; //  9216 B (per-wave diff, pad 72)
  const int t   = threadIdx.x;
  const int blk = blockIdx.x;

  // ---- stage feats tile: f32 global -> f16 LDS (HBM touched exactly once) ----
  const f32x4* src = feats4 + (size_t)blk * (64 * 64);
#pragma unroll 8
  for (int i = 0; i < 32; ++i) {
    int e = i * 128 + t;                 // 4096 float4 per tile
    int row = e >> 6, c4 = e & 63;
    f32x4 f = src[e];
    HPack p;
    p.h[0] = (_Float16)f.x; p.h[1] = (_Float16)f.y;
    p.h[2] = (_Float16)f.z; p.h[3] = (_Float16)f.w;
    *(u32x2*)&tileA[row * 264 + c4 * 4] = p.u;
  }
  __syncthreads();

  const int wave = t >> 5;
  const int lane = t & 31;               // wave32
  const int l15  = lane & 15;
  const int half = lane >> 4;
  const int rowBase = wave * 16;
  unsigned short* dW = diffB + wave * (16 * 72);

  float best[8];
#pragma unroll
  for (int j = 0; j < 8; ++j) best[j] = 3.402823e38f;

  for (int c = 0; c < NCLS; ++c) {
    const unsigned short* cw = compsT + c * (DIM * KPAD);
    const unsigned short* ci = cinv + c * (KPAD * KPAD);
    if (c + 1 < NCLS) __builtin_prefetch(compsT + (c + 1) * (DIM * KPAD), 0, 1);

    // ---- proj = feats(16xD) @ compsT(Dx64): 8 k-steps x 4 col-blocks ----
    v8f acc[4];
#pragma unroll
    for (int nb = 0; nb < 4; ++nb) {
      v8f z = {0.f,0.f,0.f,0.f,0.f,0.f,0.f,0.f};
      acc[nb] = z;
    }
#pragma unroll
    for (int db = 0; db < DIM; db += 32) {
      AFrag A;  // A 16x32 f16: lane<16 -> d = db+0..7 & db+16..23 ; lane>=16 -> +8
      const unsigned short* apn = &tileA[(rowBase + l15) * 264 + db + half * 8];
      A.q[0] = *(const u32x4*)apn;
      A.q[1] = *(const u32x4*)(apn + 16);
#pragma unroll
      for (int nb = 0; nb < 4; ++nb) {
        AFrag B;  // B 32x16 f16: lane = col (nb*16+l15), elements = rows d in table order
        const unsigned short* bp = cw + (db + half * 8) * KPAD + nb * 16 + l15;
#pragma unroll
        for (int s = 0; s < 16; ++s) { H1 x; x.u = bp[(s < 8 ? s : s + 8) * KPAD]; B.h[s] = x.h; }
        acc[nb] = __builtin_amdgcn_wmma_f32_16x16x32_f16(
            false, A.h, false, B.h, (short)0, acc[nb], false, false);
      }
    }

    // ---- diff = proj*a + b ; keep f32 in D-layout, also spill f16 to LDS ----
    v8f dv[4];
#pragma unroll
    for (int nb = 0; nb < 4; ++nb) {
      f32x2 s = ab[c * KPAD + nb * 16 + l15];
#pragma unroll
      for (int j = 0; j < 8; ++j) dv[nb][j] = acc[nb][j] * s.x + s.y;
#pragma unroll
      for (int j = 0; j < 8; ++j) {     // D-layout: VGPR j -> row j (+8 for hi half)
        H1 cv; cv.h = (_Float16)dv[nb][j];
        dW[(j + half * 8) * 72 + nb * 16 + l15] = cv.u;
      }
    }

    // ---- t = diff(16x64) @ Cinv(64x64): 2 k-steps x 4 col-blocks ----
    v8f tq[4];
#pragma unroll
    for (int nb = 0; nb < 4; ++nb) {
      v8f z = {0.f,0.f,0.f,0.f,0.f,0.f,0.f,0.f};
      tq[nb] = z;
    }
#pragma unroll
    for (int kk = 0; kk < KPAD; kk += 32) {
      AFrag A;
      const unsigned short* ap2 = &dW[l15 * 72 + kk + half * 8];
      A.q[0] = *(const u32x4*)ap2;
      A.q[1] = *(const u32x4*)(ap2 + 16);
#pragma unroll
      for (int nb = 0; nb < 4; ++nb) {
        AFrag B;
        const unsigned short* bp = ci + (kk + half * 8) * KPAD + nb * 16 + l15;
#pragma unroll
        for (int s = 0; s < 16; ++s) { H1 x; x.u = bp[(s < 8 ? s : s + 8) * KPAD]; B.h[s] = x.h; }
        tq[nb] = __builtin_amdgcn_wmma_f32_16x16x32_f16(
            false, A.h, false, B.h, (short)0, tq[nb], false, false);
      }
    }

    // ---- dist[row] = sum_l t*diff : elementwise FMA + 16-lane butterfly ----
#pragma unroll
    for (int j = 0; j < 8; ++j) {
      float p = tq[0][j] * dv[0][j] + tq[1][j] * dv[1][j]
              + tq[2][j] * dv[2][j] + tq[3][j] * dv[3][j];
      p += __shfl_xor(p, 1);
      p += __shfl_xor(p, 2);
      p += __shfl_xor(p, 4);
      p += __shfl_xor(p, 8);            // masks <16 keep the two halves separate
      best[j] = fminf(best[j], p);
    }
  }

  // lanes<16 hold rows j, lanes>=16 hold rows j+8 (replicated across the half)
  const int gm = blk * 64 + rowBase;
#pragma unroll
  for (int j = 0; j < 8; ++j) {
    if (lane == j)      out[gm + j]     = best[j];
    if (lane == 16 + j) out[gm + 8 + j] = best[j];
  }
}

extern "C" void kernel_launch(void* const* d_in, const int* in_sizes, int n_in,
                              void* d_out, int out_size, void* d_ws, size_t ws_size,
                              hipStream_t stream) {
  (void)in_sizes; (void)n_in; (void)out_size; (void)ws_size;
  const float* feats   = (const float*)d_in[0];
  const float* mean    = (const float*)d_in[1];
  const float* comps   = (const float*)d_in[2];
  const float* var     = (const float*)d_in[3];
  const float* mu      = (const float*)d_in[4];
  const float* cov_inv = (const float*)d_in[5];
  // d_in[6]=H, d_in[7]=W unused (output is flat N)

  unsigned short* compsT = (unsigned short*)d_ws;                        // 622592 B
  unsigned short* cinvW  = (unsigned short*)((char*)d_ws + COMPS_BYTES); // 155648 B
  f32x2* abW = (f32x2*)((char*)d_ws + COMPS_BYTES + CINV_BYTES);         //   9728 B

  od_prep_comps<<<NCLS * KPAD, 256, 0, stream>>>(comps, compsT);
  od_prep_cinv<<<NCLS, 256, 0, stream>>>(cov_inv, cinvW);
  od_prep_ab<<<NCLS * KPAD, 256, 0, stream>>>(mean, comps, var, mu, abW);
  od_main<<<N_PIX / 64, 128, 0, stream>>>((const f32x4*)feats, compsT, cinvW,
                                          abW, (float*)d_out);
}
